// PrefixAttention_65687229825779
// MI455X (gfx1250) — compile-verified, matrix operand order checked by reading, not run on hardware
//
#include <hip/hip_runtime.h>
#include <hip/hip_bf16.h>

// ---------------------------------------------------------------------------
// PrefixAttention for MI455X (gfx1250, wave32, WMMA).
// All GEMMs + attention run through v_wmma_f32_16x16x32_bf16 (fp32 accum).
// Panel/tile staging uses GLOBAL_LOAD_ASYNC_TO_LDS_B128 (ASYNCcnt) with
// double-buffered LDS so VMEM overlaps the WMMA pipeline.
// ---------------------------------------------------------------------------

typedef __bf16 bf16;
typedef __attribute__((ext_vector_type(16))) __bf16 bf16x16;
typedef __attribute__((ext_vector_type(8)))  __bf16 bf16x8;
typedef __attribute__((ext_vector_type(8)))  float   f32x8;

#define B_      2
#define T_      2048
#define DM_     2048
#define H_      16
#define HD_     128
#define P_      64
#define S_      2112      // P_ + T_
#define INNER_  2048
#define MROWS_  4096      // B_ * T_

// epilogue store modes for the generic GEMM
#define ST_QHEAD 1   // bf16 -> Q[b][h][t][d], scaled by 1/sqrt(HD)
#define ST_KHEAD 2   // bf16 -> K[b][h][64+t][d]
#define ST_VT    3   // bf16 -> V^T[b][h][d][64+t]
#define ST_F32   4   // f32  -> C[m][n] (final output)

static __device__ __forceinline__ f32x8 zero8() {
  f32x8 z = {0.f, 0.f, 0.f, 0.f, 0.f, 0.f, 0.f, 0.f};
  return z;
}

static __device__ __forceinline__ f32x8 wmma_bf16(bf16x16 a, bf16x16 b, f32x8 c) {
  // (neg_a, A, neg_b, B, c_mod, C, reuse_a, reuse_b)
  return __builtin_amdgcn_wmma_f32_16x16x32_bf16(false, a, false, b, (short)0, c,
                                                 false, false);
}

static __device__ __forceinline__ bf16x16 cat8(bf16x8 lo, bf16x8 hi) {
  bf16x16 r;
#pragma unroll
  for (int i = 0; i < 8; ++i) { r[i] = lo[i]; r[i + 8] = hi[i]; }
  return r;
}

// 16x32 bf16 fragment load (A layout; identical pattern serves the B operand
// when the matrix is stored N-major). Per ISA 7.12.2: lanes 0-15 hold row
// (lane&15) with k in {0..7, 16..23}; lanes 16-31 hold k in {8..15, 24..31}.
static __device__ __forceinline__ bf16x16 load_frag(const bf16* p, int row_stride,
                                                    int lane) {
  const int m  = lane & 15;
  const int hi = lane >> 4;
  const bf16* q = p + (size_t)m * row_stride + hi * 8;
  bf16x8 lo = *(const bf16x8*)(q);
  bf16x8 hv = *(const bf16x8*)(q + 16);
  return cat8(lo, hv);
}

// ---- CDNA5 async global->LDS copy (16B per lane), GVS addressing ----------
// global addr = SGPR base + 32-bit VGPR byte offset; LDS dest byte offset is
// the low 32 bits of the generic __shared__ pointer (aperture bits are high).
static __device__ __forceinline__ void async_copy_b128(const void* gbase,
                                                       unsigned byte_off,
                                                       void* lds_dst) {
  unsigned l = (unsigned)(size_t)lds_dst;
  asm volatile("global_load_async_to_lds_b128 %0, %1, %2"
               :: "v"(l), "v"(byte_off), "s"(gbase)
               : "memory");
}
static __device__ __forceinline__ void wait_async0() {
  asm volatile("s_wait_asynccnt 0x0" ::: "memory");
}

// ---------------------------------------------------------------------------
// elementwise prep kernels
// ---------------------------------------------------------------------------

__global__ void cvt_f32_bf16(const float* __restrict__ in, bf16* __restrict__ out,
                             int n) {
  int i = (blockIdx.x * blockDim.x + threadIdx.x) * 4;
  if (i < n) {
    float4 f = *(const float4*)(in + i);
    out[i + 0] = (bf16)f.x;
    out[i + 1] = (bf16)f.y;
    out[i + 2] = (bf16)f.z;
    out[i + 3] = (bf16)f.w;
  }
}

// W is (K x N) row-major fp32; produce W^T as (N x K) row-major bf16 so the
// GEMM B-operand fragment loads are contiguous per lane.
__global__ void transpose_to_bf16(const float* __restrict__ W, bf16* __restrict__ Wt,
                                  int K, int N) {
  int i = blockIdx.x * blockDim.x + threadIdx.x;   // over N*K, output-indexed
  if (i < N * K) {
    int n = i / K;
    int k = i - n * K;
    Wt[i] = (bf16)W[(size_t)k * N + n];
  }
}

// prefix_k/v are (B, P, H, D) fp32; scatter into K (b,h,s,d) and V^T (b,h,d,s)
__global__ void pack_prefix(const float* __restrict__ pk, const float* __restrict__ pv,
                            bf16* __restrict__ Kf, bf16* __restrict__ Vt) {
  int i = blockIdx.x * blockDim.x + threadIdx.x;   // B*P*H*D = 262144
  int d = i & 127;
  int h = (i >> 7) & 15;
  int p = (i >> 11) & 63;
  int b = i >> 17;
  int bh = b * H_ + h;
  Kf[((size_t)bh * S_ + p) * HD_ + d] = (bf16)pk[i];
  Vt[((size_t)bh * HD_ + d) * S_ + p] = (bf16)pv[i];
}

// ---------------------------------------------------------------------------
// Generic WMMA GEMM: C(MxN) = A(MxK) * Bt(NxK)^T, fp32 accumulate.
// 256 threads = 8 waves; block tile 128x128; wave tile 32x64; K-step 32.
// Async double-buffered LDS staging; epilogue scatters per `mode`.
// ---------------------------------------------------------------------------
__global__ __launch_bounds__(256) void gemm_bf16_wmma(
    const bf16* __restrict__ A, const bf16* __restrict__ Bt,
    bf16* __restrict__ Cb, float* __restrict__ Cf,
    int M, int N, int K, int mode, float scale) {
  __shared__ __attribute__((aligned(16))) bf16 sA[2][128 * 32];
  __shared__ __attribute__((aligned(16))) bf16 sB[2][128 * 32];

  const int tid  = threadIdx.x;
  const int lane = tid & 31;
  const int wave = tid >> 5;
  const int wm   = wave >> 1;      // 0..3 -> 32-row strip
  const int wn   = wave & 1;       // 0..1 -> 64-col strip
  const int m0   = blockIdx.y * 128;
  const int n0   = blockIdx.x * 128;

  f32x8 acc[2][4];
#pragma unroll
  for (int i = 0; i < 2; ++i)
#pragma unroll
    for (int j = 0; j < 4; ++j) acc[i][j] = zero8();

  // stage one 128x32 A panel + 128x32 B panel into LDS buffer `buf`
  auto stage = [&](int k0, int buf) {
#pragma unroll
    for (int c = 0; c < 2; ++c) {
      int cid = tid + c * 256;        // 0..511
      int r   = cid >> 2;
      int col = (cid & 3) * 8;
      async_copy_b128(A, (unsigned)(((size_t)(m0 + r) * K + k0 + col) * sizeof(bf16)),
                      &sA[buf][r * 32 + col]);
      async_copy_b128(Bt, (unsigned)(((size_t)(n0 + r) * K + k0 + col) * sizeof(bf16)),
                      &sB[buf][r * 32 + col]);
    }
  };

  stage(0, 0);
  const int nk = K / 32;
  for (int i = 0; i < nk; ++i) {
    const int buf = i & 1;
    wait_async0();                 // my panel transfers landed in LDS
    __syncthreads();               // everyone's landed; prev readers done
    if (i + 1 < nk) stage((i + 1) * 32, buf ^ 1);   // prefetch next panel

    bf16x16 af[2], bfm[4];
#pragma unroll
    for (int mi = 0; mi < 2; ++mi)
      af[mi] = load_frag(&sA[buf][(wm * 32 + mi * 16) * 32], 32, lane);
#pragma unroll
    for (int ni = 0; ni < 4; ++ni)
      bfm[ni] = load_frag(&sB[buf][(wn * 64 + ni * 16) * 32], 32, lane);

#pragma unroll
    for (int mi = 0; mi < 2; ++mi)
#pragma unroll
      for (int ni = 0; ni < 4; ++ni)
        acc[mi][ni] = wmma_bf16(af[mi], bfm[ni], acc[mi][ni]);
  }

  const int hi = lane >> 4;
  const int nl = lane & 15;
#pragma unroll
  for (int mi = 0; mi < 2; ++mi) {
#pragma unroll
    for (int ni = 0; ni < 4; ++ni) {
#pragma unroll
      for (int v = 0; v < 8; ++v) {
        int m = m0 + wm * 32 + mi * 16 + v + 8 * hi;   // C row (ISA C layout)
        int n = n0 + wn * 64 + ni * 16 + nl;           // C col
        float val = acc[mi][ni][v] * scale;
        if (mode == ST_F32) {
          Cf[(size_t)m * N + n] = val;
        } else {
          int b = m >> 11;        // m / T_
          int t = m & (T_ - 1);
          int h = n >> 7;         // n / HD_
          int d = n & (HD_ - 1);
          size_t idx;
          if (mode == ST_QHEAD)
            idx = ((size_t)(b * H_ + h) * T_ + t) * HD_ + d;
          else if (mode == ST_KHEAD)
            idx = ((size_t)(b * H_ + h) * S_ + (P_ + t)) * HD_ + d;
          else  // ST_VT : transposed V so PV-GEMM B-operand is contiguous
            idx = ((size_t)(b * H_ + h) * HD_ + d) * S_ + (P_ + t);
          Cb[idx] = (bf16)val;
        }
      }
    }
  }
}

// ---------------------------------------------------------------------------
// Flash attention, one (b,h, 128-query tile) per block; 8 waves x 16 rows.
// K/V chunks (32 keys) staged once per block via async-to-LDS (shared by all
// 8 waves), double-buffered. Q pre-scaled by 1/sqrt(HD); online softmax.
// LDS: 2*8KB (K) + 2*8KB (V) + 8KB (P tiles) = 40KB / block.
// ---------------------------------------------------------------------------
__global__ __launch_bounds__(256) void attn_wmma(
    const bf16* __restrict__ Qh, const bf16* __restrict__ Kf,
    const bf16* __restrict__ Vt, bf16* __restrict__ attn) {
  __shared__ __attribute__((aligned(16))) bf16 sK[2][32 * HD_];
  __shared__ __attribute__((aligned(16))) bf16 sV[2][HD_ * 32];
  __shared__ __attribute__((aligned(16))) bf16 sP[8][16 * 32];

  const int blk  = blockIdx.x;        // B_*H_*(T_/128) = 512
  const int qt   = blk & 15;
  const int bh   = blk >> 4;          // 0..31
  const int b    = bh >> 4;
  const int h    = bh & 15;
  const int tid  = threadIdx.x;
  const int wave = tid >> 5;
  const int lane = tid & 31;
  const int hi   = lane >> 4;
  const int nl   = lane & 15;
  const int t0   = qt * 128 + wave * 16;

  const bf16* Qbase = Qh + ((size_t)bh * T_ + t0) * HD_;
  const bf16* Kbase = Kf + (size_t)bh * S_ * HD_;
  const bf16* Vbase = Vt + (size_t)bh * HD_ * S_;
  bf16* myP = &sP[wave][0];

  // stage one 32x128 K chunk and 128x32 V^T chunk into LDS buffer `buf`
  auto stageKV = [&](int s0, int buf) {
#pragma unroll
    for (int c = 0; c < 2; ++c) {
      int cid = tid + c * 256;          // 0..511
      int kr  = cid >> 4;               // K row (s-local), 16 chunks/row
      int kc  = (cid & 15) * 8;
      async_copy_b128(Kbase,
                      (unsigned)((((size_t)(s0 + kr)) * HD_ + kc) * sizeof(bf16)),
                      &sK[buf][kr * HD_ + kc]);
      int vr = cid >> 2;                // V^T row (d), 4 chunks/row
      int vc = (cid & 3) * 8;
      async_copy_b128(Vbase,
                      (unsigned)((((size_t)vr) * S_ + s0 + vc) * sizeof(bf16)),
                      &sV[buf][vr * 32 + vc]);
    }
  };

  bf16x16 qf[4];
#pragma unroll
  for (int kd = 0; kd < 4; ++kd) qf[kd] = load_frag(Qbase + kd * 32, HD_, lane);

  f32x8 o[8];
#pragma unroll
  for (int f = 0; f < 8; ++f) o[f] = zero8();
  float mrun[8], lrun[8];
#pragma unroll
  for (int v = 0; v < 8; ++v) { mrun[v] = -1e30f; lrun[v] = 0.f; }

  stageKV(0, 0);
  const int niter = S_ / 32;            // 66
  for (int it = 0; it < niter; ++it) {
    const int buf = it & 1;
    wait_async0();
    __syncthreads();
    if (it + 1 < niter) stageKV((it + 1) * 32, buf ^ 1);

    // logits tile: 16 rows x 32 keys = two C fragments, fed from LDS
    f32x8 a0 = zero8(), a1 = zero8();
#pragma unroll
    for (int kd = 0; kd < 4; ++kd) {
      bf16x16 kb0 = load_frag(&sK[buf][0 * HD_ + kd * 32], HD_, lane);
      bf16x16 kb1 = load_frag(&sK[buf][16 * HD_ + kd * 32], HD_, lane);
      a0 = wmma_bf16(qf[kd], kb0, a0);
      a1 = wmma_bf16(qf[kd], kb1, a1);
    }

    // online softmax: row r = v + 8*hi lives on the 16 lanes sharing `hi`
    float alpha[8];
#pragma unroll
    for (int v = 0; v < 8; ++v) {
      float v0 = a0[v], v1 = a1[v];
      float cm = fmaxf(v0, v1);
#pragma unroll
      for (int md = 1; md < 16; md <<= 1) cm = fmaxf(cm, __shfl_xor(cm, md, 16));
      float mn = fmaxf(mrun[v], cm);
      float al = __expf(mrun[v] - mn);
      float p0 = __expf(v0 - mn);
      float p1 = __expf(v1 - mn);
      float rs = p0 + p1;
#pragma unroll
      for (int md = 1; md < 16; md <<= 1) rs += __shfl_xor(rs, md, 16);
      lrun[v] = lrun[v] * al + rs;
      mrun[v] = mn;
      alpha[v] = al;
      int r = v + 8 * hi;
      myP[r * 32 + nl]      = (bf16)p0;   // C layout -> LDS (row-major 16x32)
      myP[r * 32 + 16 + nl] = (bf16)p1;
    }
#pragma unroll
    for (int f = 0; f < 8; ++f)
#pragma unroll
      for (int v = 0; v < 8; ++v) o[f][v] *= alpha[v];

    // wave-local LDS round-trip: ensure DS stores landed before A-layout reload
    asm volatile("s_wait_dscnt 0x0" ::: "memory");
    __builtin_amdgcn_wave_barrier();
    bf16x16 pf = load_frag(myP, 32, lane);

    // O += P(16x32) x V(32x128); V^T rows contiguous in s inside LDS
#pragma unroll
    for (int f = 0; f < 8; ++f) {
      bf16x16 vf = load_frag(&sV[buf][(f * 16) * 32], 32, lane);
      o[f] = wmma_bf16(pf, vf, o[f]);
    }
  }

  // normalize and store row-major (B*T, INNER) for the output projection
#pragma unroll
  for (int v = 0; v < 8; ++v) {
    int r = v + 8 * hi;
    int t = t0 + r;
    float inv = 1.0f / lrun[v];
#pragma unroll
    for (int f = 0; f < 8; ++f) {
      int d = f * 16 + nl;
      attn[((size_t)(b * T_ + t)) * INNER_ + h * HD_ + d] = (bf16)(o[f][v] * inv);
    }
  }
}

// ---------------------------------------------------------------------------
// host launcher
// ---------------------------------------------------------------------------
extern "C" void kernel_launch(void* const* d_in, const int* in_sizes, int n_in,
                              void* d_out, int out_size, void* d_ws, size_t ws_size,
                              hipStream_t stream) {
  (void)in_sizes; (void)n_in; (void)out_size; (void)ws_size;

  const float* x  = (const float*)d_in[0];
  const float* pk = (const float*)d_in[1];
  const float* pv = (const float*)d_in[2];
  const float* Wq = (const float*)d_in[3];
  const float* Wk = (const float*)d_in[4];
  const float* Wv = (const float*)d_in[5];
  const float* Wo = (const float*)d_in[6];
  float* out = (float*)d_out;

  // workspace carve-up (~102 MB; attn aliases xb which is dead after V GEMM)
  char* ws = (char*)d_ws;
  size_t off = 0;
  auto carve = [&](size_t bytes) -> void* {
    void* p = ws + off;
    off += (bytes + 255) & ~(size_t)255;
    return p;
  };
  bf16* xb    = (bf16*)carve((size_t)MROWS_ * DM_ * sizeof(bf16));   // 16.8 MB
  bf16* WqT   = (bf16*)carve((size_t)DM_ * INNER_ * sizeof(bf16));
  bf16* WkT   = (bf16*)carve((size_t)DM_ * INNER_ * sizeof(bf16));
  bf16* WvT   = (bf16*)carve((size_t)DM_ * INNER_ * sizeof(bf16));
  bf16* WoT   = (bf16*)carve((size_t)INNER_ * DM_ * sizeof(bf16));
  bf16* Qhd   = (bf16*)carve((size_t)B_ * H_ * T_ * HD_ * sizeof(bf16));
  bf16* Kfull = (bf16*)carve((size_t)B_ * H_ * S_ * HD_ * sizeof(bf16));
  bf16* Vtr   = (bf16*)carve((size_t)B_ * H_ * HD_ * S_ * sizeof(bf16));
  bf16* attnb = xb;  // alias: x-bf16 is dead once the V projection finished

  // 1) convert x to bf16
  {
    int n = MROWS_ * DM_;
    cvt_f32_bf16<<<n / 1024, 256, 0, stream>>>(x, xb, n);
  }
  // 2) transpose+convert weights to N-major bf16
  {
    int n = DM_ * INNER_;
    int g = (n + 255) / 256;
    transpose_to_bf16<<<g, 256, 0, stream>>>(Wq, WqT, DM_, INNER_);
    transpose_to_bf16<<<g, 256, 0, stream>>>(Wk, WkT, DM_, INNER_);
    transpose_to_bf16<<<g, 256, 0, stream>>>(Wv, WvT, DM_, INNER_);
    transpose_to_bf16<<<g, 256, 0, stream>>>(Wo, WoT, INNER_, DM_);
  }
  // 3) QKV projections (scatter straight into attention layouts)
  {
    dim3 gg(INNER_ / 128, MROWS_ / 128);  // (16, 32)
    const float qscale = 0.0883883476483184f;  // 1/sqrt(128)
    gemm_bf16_wmma<<<gg, 256, 0, stream>>>(xb, WqT, Qhd, nullptr,
                                           MROWS_, INNER_, DM_, ST_QHEAD, qscale);
    gemm_bf16_wmma<<<gg, 256, 0, stream>>>(xb, WkT, Kfull, nullptr,
                                           MROWS_, INNER_, DM_, ST_KHEAD, 1.0f);
    gemm_bf16_wmma<<<gg, 256, 0, stream>>>(xb, WvT, Vtr, nullptr,
                                           MROWS_, INNER_, DM_, ST_VT, 1.0f);
  }
  // 4) prefix K/V into the head layouts
  pack_prefix<<<(B_ * P_ * H_ * HD_) / 256, 256, 0, stream>>>(pk, pv, Kfull, Vtr);
  // 5) flash attention
  attn_wmma<<<B_ * H_ * (T_ / 128), 256, 0, stream>>>(Qhd, Kfull, Vtr, attnb);
  // 6) output projection -> fp32 d_out
  {
    dim3 go(DM_ / 128, MROWS_ / 128);
    gemm_bf16_wmma<<<go, 256, 0, stream>>>(attnb, WoT, nullptr, out,
                                           MROWS_, DM_, INNER_, ST_F32, 1.0f);
  }
}
